// InterFrequencyAttention_20452634263752
// MI455X (gfx1250) — compile-verified, matrix operand order checked by reading, not run on hardware
//
#include <hip/hip_runtime.h>

// ---------------------------------------------------------------------------
// InterFrequencyAttention for MI455X (gfx1250), fused via 64x64 Gram matrices.
//   b=4, c=64, L=n*h*w=102400, dims=192
//   G_ll = LF LF^T, G_lh = LF HF^T, G_hh = HF HF^T   (per batch, 64x64)
//   S    = Wq G_lh Wk^T ; nq^2 = diag(Wq G_ll Wq^T) ; nk^2 = diag(Wk G_hh Wk^T)
//   A    = softmax(S / (nq nk)) ; M = A Wv (192x64) ; out = M HF (192xL)
// Heavy passes use v_wmma_f32_16x16x32_bf16 (f32 accumulate).
// ---------------------------------------------------------------------------

#define B_SZ   4
#define C_SZ   64
#define L_TOT  102400
#define DIMS   192
#define NWG_L  100         // gram workgroups per batch (400 WGs total)
#define SPAN   1024        // L per gram WG  (100*1024 = 102400)
#define CHUNK  128         // L per LDS stage
#define NCHUNK 8           // SPAN / CHUNK
#define OUTCH  256         // L per out WG
#define NOUTWG 400         // L_TOT / OUTCH

typedef __bf16  v16bf __attribute__((ext_vector_type(16)));
typedef float   v8f   __attribute__((ext_vector_type(8)));
typedef float   f32x4 __attribute__((ext_vector_type(4)));
typedef unsigned int u32x4 __attribute__((ext_vector_type(4)));
typedef unsigned int u32x2 __attribute__((ext_vector_type(2)));

union BfVec { u32x4 q[2]; v16bf v; };
static_assert(sizeof(BfVec) == 32, "BfVec size");

__device__ __forceinline__ unsigned int f2bf(float f) {
    unsigned int u = __float_as_uint(f);
    return (u + 0x7fffu + ((u >> 16) & 1u)) >> 16;   // round-to-nearest-even
}
__device__ __forceinline__ unsigned int pack2(float lo, float hi) {
    return f2bf(lo) | (f2bf(hi) << 16);
}
__device__ __forceinline__ v8f wmma_bf16(const BfVec& a, const BfVec& b, v8f c) {
    return __builtin_amdgcn_wmma_f32_16x16x32_bf16(false, a.v, false, b.v,
                                                   (short)0, c, false, false);
}

// ---------------------------------------------------------------------------
// Kernel 1: per-WG partial Gram matrices G_ll, G_lh, G_hh over a 1024-wide
// slab of L. 16 waves, each owns one (ti,tj) 16x16 tile of all three Grams.
// LF/HF chunk staged to LDS as bf16 in WMMA-ready row-major runs.
// part layout: [b][wg][3][64][64] f32
// ---------------------------------------------------------------------------
__global__ __launch_bounds__(512) void gram_partial_kernel(
    const float* __restrict__ lf, const float* __restrict__ hf,
    float* __restrict__ part)
{
    __shared__ alignas(16) unsigned short LFs[64][CHUNK];
    __shared__ alignas(16) unsigned short HFs[64][CHUNK];

    const int b    = blockIdx.y;
    const int wg   = blockIdx.x;
    const int t    = threadIdx.x;
    const int wave = t >> 5;
    const int lane = t & 31;
    const int g    = lane >> 4;      // half-wave (K-group)
    const int mn   = lane & 15;      // A row / B col within tile
    const int ti   = wave >> 2;
    const int tj   = wave & 3;

    const float* lfb = lf + (size_t)b * C_SZ * L_TOT;
    const float* hfb = hf + (size_t)b * C_SZ * L_TOT;
    const size_t lbase = (size_t)wg * SPAN;

    // staging role
    const int sr  = t >> 3;          // row 0..63
    const int sl0 = (t & 7) * 4;     // col base

    v8f cll = {}; v8f clh = {}; v8f chh = {};

    for (int ch = 0; ch < NCHUNK; ++ch) {
        const size_t l0 = lbase + (size_t)ch * CHUNK;
        __syncthreads();
        #pragma unroll
        for (int p = 0; p < 4; ++p) {
            const int lj = sl0 + p * 32;
            f32x4 a = *(const f32x4*)(lfb + (size_t)sr * L_TOT + l0 + lj);
            f32x4 c = *(const f32x4*)(hfb + (size_t)sr * L_TOT + l0 + lj);
            u32x2 pa = { pack2(a.x, a.y), pack2(a.z, a.w) };
            u32x2 pc = { pack2(c.x, c.y), pack2(c.z, c.w) };
            *(u32x2*)&LFs[sr][lj] = pa;
            *(u32x2*)&HFs[sr][lj] = pc;
        }
        __syncthreads();

        const unsigned short* arow_lf = &LFs[ti * 16 + mn][0];
        const unsigned short* arow_hf = &HFs[ti * 16 + mn][0];
        const unsigned short* brow_lf = &LFs[tj * 16 + mn][0];
        const unsigned short* brow_hf = &HFs[tj * 16 + mn][0];

        #pragma unroll
        for (int kb = 0; kb < CHUNK; kb += 32) {
            const int k0 = kb + g * 8;
            const int k1 = kb + 16 + g * 8;
            BfVec alf, ahf, blf, bhf;
            alf.q[0] = *(const u32x4*)(arow_lf + k0); alf.q[1] = *(const u32x4*)(arow_lf + k1);
            ahf.q[0] = *(const u32x4*)(arow_hf + k0); ahf.q[1] = *(const u32x4*)(arow_hf + k1);
            blf.q[0] = *(const u32x4*)(brow_lf + k0); blf.q[1] = *(const u32x4*)(brow_lf + k1);
            bhf.q[0] = *(const u32x4*)(brow_hf + k0); bhf.q[1] = *(const u32x4*)(brow_hf + k1);
            cll = wmma_bf16(alf, blf, cll);
            clh = wmma_bf16(alf, bhf, clh);
            chh = wmma_bf16(ahf, bhf, chh);
        }
    }

    // C/D layout: VGPR v -> row = v + 8*g, col = mn (within 16x16 tile)
    float* pb = part + (((size_t)b * NWG_L + wg) * 3) * 4096;
    const int row0 = ti * 16 + g * 8;
    const int col  = tj * 16 + mn;
    #pragma unroll
    for (int v = 0; v < 8; ++v) {
        pb[0 * 4096 + (row0 + v) * 64 + col] = cll[v];
        pb[1 * 4096 + (row0 + v) * 64 + col] = clh[v];
        pb[2 * 4096 + (row0 + v) * 64 + col] = chh[v];
    }
}

// ---------------------------------------------------------------------------
// Kernel 2: deterministic fixed-order reduction of partials -> G[b][3][64][64]
// ---------------------------------------------------------------------------
__global__ __launch_bounds__(256) void gram_reduce_kernel(
    const float* __restrict__ part, float* __restrict__ G)
{
    const int idx = blockIdx.x * 256 + threadIdx.x;     // 0 .. 4*12288-1
    const int b = idx / 12288;
    const int r = idx % 12288;
    const float* p = part + (size_t)b * NWG_L * 12288 + r;
    float s = 0.0f;
    for (int wg = 0; wg < NWG_L; ++wg) s += p[(size_t)wg * 12288];
    G[idx] = s;
}

// ---------------------------------------------------------------------------
// Kernel 3: norms + softmax + M = att * Wv.  One block per batch, thread d
// owns attention row d.  Flash-style single pass over e (no S materialized).
// ---------------------------------------------------------------------------
__global__ __launch_bounds__(192) void attn_mid_kernel(
    const float* __restrict__ wq, const float* __restrict__ wkv,
    const float* __restrict__ G, float* __restrict__ M)
{
    __shared__ float Gll[64][64];
    __shared__ float Glh[64][64];
    __shared__ float Ghh[64][64];
    __shared__ float nkinv[DIMS];

    const int b = blockIdx.x;
    const int d = threadIdx.x;
    const float* Gb = G + (size_t)b * 3 * 4096;

    for (int i = d; i < 4096; i += DIMS) {
        (&Gll[0][0])[i] = Gb[i];
        (&Glh[0][0])[i] = Gb[4096 + i];
        (&Ghh[0][0])[i] = Gb[8192 + i];
    }
    __syncthreads();

    const float* wqd = wq  + (size_t)d * 64;
    const float* wkd = wkv + (size_t)d * 64;     // K rows = wkv[0..191]

    float t1[64];

    // nk^2[d] = wkd * Ghh * wkd^T
    #pragma unroll
    for (int c2 = 0; c2 < 64; ++c2) t1[c2] = 0.0f;
    for (int c = 0; c < 64; ++c) {
        const float wc = wkd[c];
        #pragma unroll
        for (int c2 = 0; c2 < 64; ++c2) t1[c2] += wc * Ghh[c][c2];
    }
    float nk2 = 0.0f;
    #pragma unroll
    for (int c2 = 0; c2 < 64; ++c2) nk2 += t1[c2] * wkd[c2];
    nkinv[d] = 1.0f / fmaxf(sqrtf(fmaxf(nk2, 0.0f)), 1e-12f);

    // nq^2[d] = wqd * Gll * wqd^T
    #pragma unroll
    for (int c2 = 0; c2 < 64; ++c2) t1[c2] = 0.0f;
    for (int c = 0; c < 64; ++c) {
        const float wc = wqd[c];
        #pragma unroll
        for (int c2 = 0; c2 < 64; ++c2) t1[c2] += wc * Gll[c][c2];
    }
    float nq2 = 0.0f;
    #pragma unroll
    for (int c2 = 0; c2 < 64; ++c2) nq2 += t1[c2] * wqd[c2];
    const float rq = 1.0f / fmaxf(sqrtf(fmaxf(nq2, 0.0f)), 1e-12f);

    // t1 = wqd * Glh  (row d of Wq G_lh)
    #pragma unroll
    for (int c2 = 0; c2 < 64; ++c2) t1[c2] = 0.0f;
    for (int c = 0; c < 64; ++c) {
        const float wc = wqd[c];
        #pragma unroll
        for (int c2 = 0; c2 < 64; ++c2) t1[c2] += wc * Glh[c][c2];
    }
    __syncthreads();

    // online softmax over e with fused V (= Wv row) accumulation
    float mx = -1e30f, sum = 0.0f;
    float vacc[64];
    #pragma unroll
    for (int c = 0; c < 64; ++c) vacc[c] = 0.0f;

    for (int e = 0; e < DIMS; ++e) {
        const float* wke = wkv + (size_t)e * 64;
        const float* wve = wkv + (size_t)(DIMS + e) * 64;
        float s = 0.0f;
        #pragma unroll
        for (int c = 0; c < 64; ++c) s += t1[c] * wke[c];
        s *= rq * nkinv[e];
        const float nm   = fmaxf(mx, s);
        const float corr = __expf(mx - nm);
        const float p    = __expf(s - nm);
        sum = sum * corr + p;
        #pragma unroll
        for (int c = 0; c < 64; ++c) vacc[c] = vacc[c] * corr + p * wve[c];
        mx = nm;
    }
    const float inv = 1.0f / sum;
    float* Md = M + ((size_t)b * DIMS + d) * 64;
    #pragma unroll
    for (int c = 0; c < 64; ++c) Md[c] = vacc[c] * inv;
}

// ---------------------------------------------------------------------------
// Kernel 4: out = M (192x64) * HF (64xL), bf16 WMMA, f32 out.
// Per WG: 256 columns of L.  HF chunk staged TRANSPOSED to LDS (bf16) so
// B-operand reads are contiguous 16B runs; M staged as bf16 row-major.
// Each of the 8 waves runs two 16-column tiles x 12 row tiles (48 WMMAs).
// ---------------------------------------------------------------------------
__global__ __launch_bounds__(256) void out_gemm_kernel(
    const float* __restrict__ hf, const float* __restrict__ M,
    float* __restrict__ out)
{
    __shared__ alignas(16) unsigned short Ms[DIMS][64];   // 24 KB
    __shared__ alignas(16) unsigned short HFt[OUTCH][64]; // 32 KB (l-major)

    const int b    = blockIdx.y;
    const size_t l0 = (size_t)blockIdx.x * OUTCH;
    const int t    = threadIdx.x;
    const int wave = t >> 5;
    const int lane = t & 31;
    const int g    = lane >> 4;
    const int mn   = lane & 15;

    const float* hfb = hf + (size_t)b * C_SZ * L_TOT;
    const float* Mb  = M  + (size_t)b * DIMS * 64;
    float*       ob  = out + (size_t)b * DIMS * L_TOT;

    // stage M (192*64 = 12288 f32 -> bf16): 256 threads x 12 float4
    #pragma unroll
    for (int i = 0; i < 12; ++i) {
        const int e4 = (t + i * 256) * 4;
        f32x4 a = *(const f32x4*)(Mb + e4);
        u32x2 pa = { pack2(a.x, a.y), pack2(a.z, a.w) };
        *(u32x2*)&(((unsigned short*)Ms)[e4]) = pa;
    }
    // stage HF transposed: thread -> row c = t>>2, 4 consecutive l per pass
    {
        const int c = t >> 2;
        #pragma unroll
        for (int p = 0; p < 16; ++p) {
            const int lj = (t & 3) * 4 + p * 16;
            f32x4 a = *(const f32x4*)(hfb + (size_t)c * L_TOT + l0 + lj);
            HFt[lj + 0][c] = (unsigned short)f2bf(a.x);
            HFt[lj + 1][c] = (unsigned short)f2bf(a.y);
            HFt[lj + 2][c] = (unsigned short)f2bf(a.z);
            HFt[lj + 3][c] = (unsigned short)f2bf(a.w);
        }
    }
    __syncthreads();

    // two column tiles per wave: ct = wave, wave+8
    #pragma unroll
    for (int cti = 0; cti < 2; ++cti) {
        const int ct = wave + cti * 8;
        // B operand: lane(n=mn,g) holds column l0+ct*16+mn, K runs over c
        const unsigned short* brow = &HFt[ct * 16 + mn][0];
        BfVec b0, b1;
        b0.q[0] = *(const u32x4*)(brow +  0 + g * 8);
        b0.q[1] = *(const u32x4*)(brow + 16 + g * 8);
        b1.q[0] = *(const u32x4*)(brow + 32 + g * 8);
        b1.q[1] = *(const u32x4*)(brow + 48 + g * 8);

        const size_t colAddr = l0 + (size_t)ct * 16 + mn;
        #pragma unroll
        for (int dt = 0; dt < 12; ++dt) {
            const unsigned short* arow = &Ms[dt * 16 + mn][0];
            BfVec a0, a1;
            a0.q[0] = *(const u32x4*)(arow +  0 + g * 8);
            a0.q[1] = *(const u32x4*)(arow + 16 + g * 8);
            a1.q[0] = *(const u32x4*)(arow + 32 + g * 8);
            a1.q[1] = *(const u32x4*)(arow + 48 + g * 8);
            v8f acc = {};
            acc = wmma_bf16(a0, b0, acc);
            acc = wmma_bf16(a1, b1, acc);
            #pragma unroll
            for (int v = 0; v < 8; ++v)
                ob[(size_t)(dt * 16 + g * 8 + v) * L_TOT + colAddr] = acc[v];
        }
    }
}

// ---------------------------------------------------------------------------
extern "C" void kernel_launch(void* const* d_in, const int* in_sizes, int n_in,
                              void* d_out, int out_size, void* d_ws, size_t ws_size,
                              hipStream_t stream) {
    const float* lf  = (const float*)d_in[0];
    const float* hf  = (const float*)d_in[1];
    const float* wq  = (const float*)d_in[2];
    const float* wkv = (const float*)d_in[3];
    float* out = (float*)d_out;

    float* part = (float*)d_ws;                               // 4*100*3*4096 f32
    float* G    = part + (size_t)B_SZ * NWG_L * 3 * 4096;     // 4*3*4096  f32
    float* Mm   = G    + (size_t)B_SZ * 3 * 4096;             // 4*192*64  f32

    gram_partial_kernel<<<dim3(NWG_L, B_SZ), 512, 0, stream>>>(lf, hf, part);
    gram_reduce_kernel<<<(B_SZ * 3 * 4096) / 256, 256, 0, stream>>>(part, G);
    attn_mid_kernel<<<B_SZ, DIMS, 0, stream>>>(wq, wkv, G, Mm);
    out_gemm_kernel<<<dim3(NOUTWG, B_SZ), 256, 0, stream>>>(hf, Mm, out);
}
